// CGRE_78434692759889
// MI455X (gfx1250) — matrix operand order repaired and checked by reading, not compiled, not run on hardware
//
#include <hip/hip_runtime.h>

typedef __attribute__((ext_vector_type(16))) _Float16 v16h;
typedef __attribute__((ext_vector_type(8)))  _Float16 v8h;
typedef __attribute__((ext_vector_type(4)))  _Float16 v4h;
typedef __attribute__((ext_vector_type(8)))  float    v8f;
typedef __attribute__((ext_vector_type(4)))  float    v4f;

#define NROWS 65536
#define NBAGS 8192
#define NC    1024
#define NR    100

// ---------------------------------------------------------------------------
// Kernel 0: convert W (R x C, f32) -> f16 for the WMMA GEMM. 102400 elements.
// ---------------------------------------------------------------------------
__global__ __launch_bounds__(256) void cgre_convW(const float* __restrict__ W,
                                                  _Float16* __restrict__ Wh,
                                                  int n) {
    int i = blockIdx.x * 256 + threadIdx.x;
    if (i < n) Wh[i] = (_Float16)W[i];
}

// ---------------------------------------------------------------------------
// Kernel 1: fused online-softmax bag pooling. One wave (32 lanes) per bag.
// Each bag is a contiguous row range [start,end) of X, so the wave streams a
// contiguous block of memory with coalesced b128 loads. X is read exactly
// once (256 MB total -> the bandwidth floor of this problem). Since X
// (256 MB) exceeds L2 (192 MB) and is never re-read, its loads are marked
// non-temporal so they don't evict the L2-resident bag/W data that kernel 2
// depends on. Lane L owns float4 chunks {L, L+32, ..., L+224} of each row.
// ---------------------------------------------------------------------------
__global__ __launch_bounds__(256) void cgre_bag_softmax(
    const float* __restrict__ X,
    const float* __restrict__ Cons,
    const int*   __restrict__ scope,   // (B,2) start/end
    const int*   __restrict__ rel,     // (B,)
    _Float16*    __restrict__ bagOut)  // (B, C) f16 row-major
{
    const int wave = threadIdx.x >> 5;
    const int lane = threadIdx.x & 31;
    const int bag  = blockIdx.x * 8 + wave;
    if (bag >= NBAGS) return;

    const int start = scope[2 * bag + 0];
    const int end   = scope[2 * bag + 1];

    // Per-bag constraint vector, kept in registers (32 floats / lane).
    // Constraints is only 400 KB and shared across bags -> regular (cached).
    const v4f* conp = (const v4f*)(Cons + (size_t)rel[bag] * NC);
    v4f con[8];
#pragma unroll
    for (int k = 0; k < 8; ++k) con[k] = conp[lane + 32 * k];

    v4f acc[8];
#pragma unroll
    for (int k = 0; k < 8; ++k) acc[k] = (v4f){0.f, 0.f, 0.f, 0.f};

    float m = -INFINITY;
    float s = 0.f;

    for (int row = start; row < end; ++row) {
        const v4f* xp = (const v4f*)(X + (size_t)row * NC);
        v4f x[8];
#pragma unroll
        for (int k = 0; k < 8; ++k)
            x[k] = __builtin_nontemporal_load(xp + lane + 32 * k);

        // lane-local partial dot (32 elements)
        float part = 0.f;
#pragma unroll
        for (int k = 0; k < 8; ++k) {
            part += x[k].x * con[k].x + x[k].y * con[k].y +
                    x[k].z * con[k].z + x[k].w * con[k].w;
        }
        // wave32 butterfly reduction -> full dot in every lane
#pragma unroll
        for (int off = 16; off >= 1; off >>= 1)
            part += __shfl_xor(part, off, 32);

        // online softmax update with rescale
        const float mn    = fmaxf(m, part);
        const float scale = __expf(m - mn);    // exp(-inf)=0 on first iter
        const float p     = __expf(part - mn);
        s = s * scale + p;
#pragma unroll
        for (int k = 0; k < 8; ++k)
            acc[k] = acc[k] * scale + p * x[k];
        m = mn;
    }

    const float inv = (s > 0.f) ? (1.f / s) : 0.f;
    _Float16* outp = bagOut + (size_t)bag * NC;
#pragma unroll
    for (int k = 0; k < 8; ++k) {
        v4f v = acc[k] * inv;
        v4h h;
        h.x = (_Float16)v.x;
        h.y = (_Float16)v.y;
        h.z = (_Float16)v.z;
        h.w = (_Float16)v.w;
        *(v4h*)(outp + 4 * (size_t)(lane + 32 * k)) = h;   // RT: re-read by k2
    }
}

// ---------------------------------------------------------------------------
// Kernel 2: out(8192x100) = bag(8192x1024) @ W^T + b via v_wmma_f32_16x16x32_f16.
// One wave per 16-row M-tile; each wave computes all 7 N-tiles (100 -> 7x16,
// masked at store) reusing the A fragment 7x per K-step. Both operands are
// L2-resident (bag 16 MB, W 200 KB).
//
// Fragment layouts used by this code (16-bit, wave32):
//   A (16x32 MxK): lane l holds row M=l&15; v16h element j holds
//                  K = ((l>>4)*16) + j                   -> two 16B loads
//   B (32x16 KxN): lane l holds col N=l&15; element j holds
//                  K = ((l>>4)*16) + j                   -> one 32B load
//   C/D (16x16 f32): VGPR i, lanes 0-15 -> M=i, lanes 16-31 -> M=8+i; N=l&15
// ---------------------------------------------------------------------------
__global__ __launch_bounds__(256) void cgre_gemm_wmma(
    const _Float16* __restrict__ bag,   // (8192,1024) f16
    const _Float16* __restrict__ Wh,    // (100,1024)  f16
    const float*    __restrict__ bias,  // (100,)
    float*          __restrict__ out)   // (8192,100)
{
    const int wave  = threadIdx.x >> 5;
    const int lane  = threadIdx.x & 31;
    const int mtile = blockIdx.x * 8 + wave;       // 512 M-tiles
    if (mtile >= NBAGS / 16) return;

    const int laneN = lane & 15;
    const int hi    = lane >> 4;

    const _Float16* arow = bag + (size_t)(mtile * 16 + laneN) * NC;

    v8f acc[7];
#pragma unroll
    for (int n = 0; n < 7; ++n) acc[n] = {};

    for (int k0 = 0; k0 < NC; k0 += 32) {
        // A fragment: K in {k0+hi*16 .. +15} (contiguous, matches B)
        v8h alo = *(const v8h*)(arow + k0 + hi * 16);
        v8h ahi = *(const v8h*)(arow + k0 + hi * 16 + 8);
        v16h a;
#pragma unroll
        for (int j = 0; j < 8; ++j) { a[j] = alo[j]; a[8 + j] = ahi[j]; }

#pragma unroll
        for (int n = 0; n < 7; ++n) {
            int r = n * 16 + laneN;
            if (r > NR - 1) r = NR - 1;            // clamp; masked at store
            v16h bb = *(const v16h*)(Wh + (size_t)r * NC + k0 + hi * 16);
            acc[n] = __builtin_amdgcn_wmma_f32_16x16x32_f16(
                false, a, false, bb, (short)0, acc[n], false, false);
        }
    }

    const int mbase = mtile * 16 + hi * 8;
#pragma unroll
    for (int n = 0; n < 7; ++n) {
        const int r = n * 16 + laneN;
        if (r < NR) {
            const float bv = bias[r];
#pragma unroll
            for (int i = 0; i < 8; ++i)
                out[(size_t)(mbase + i) * NR + r] = acc[n][i] + bv;
        }
    }
}

// ---------------------------------------------------------------------------
// Host-side launcher
// ---------------------------------------------------------------------------
extern "C" void kernel_launch(void* const* d_in, const int* in_sizes, int n_in,
                              void* d_out, int out_size, void* d_ws, size_t ws_size,
                              hipStream_t stream) {
    const float* X      = (const float*)d_in[0];   // (65536,1024)
    const float* Cons   = (const float*)d_in[1];   // (100,1024)
    const float* W      = (const float*)d_in[2];   // (100,1024)
    const float* b      = (const float*)d_in[3];   // (100,)
    const int*   scope  = (const int*)d_in[4];     // (8192,2)
    const int*   rel    = (const int*)d_in[5];     // (8192,)
    float*       out    = (float*)d_out;           // (8192,100)

    // workspace layout: bag f16 (8192*1024), then W f16 (100*1024)
    _Float16* bagH = (_Float16*)d_ws;
    _Float16* Wh   = bagH + (size_t)NBAGS * NC;

    cgre_convW<<<(NR * NC + 255) / 256, 256, 0, stream>>>(W, Wh, NR * NC);
    cgre_bag_softmax<<<NBAGS / 8, 256, 0, stream>>>(X, Cons, scope, rel, bagH);
    cgre_gemm_wmma<<<(NBAGS / 16) / 8, 256, 0, stream>>>(bagH, Wh, b, out);
}
